// snnBlock_8546984919494
// MI455X (gfx1250) — compile-verified, matrix operand order checked by reading, not run on hardware
//
#include <hip/hip_runtime.h>
#include <math.h>

typedef __attribute__((ext_vector_type(16))) __bf16 v16bf;
typedef __attribute__((ext_vector_type(8)))  float  v8f;

#define T_STEPS 4
#define BATCH   8
#define CIN     64
#define COUT    128
#define HH      48
#define WW      48
#define HP      50
#define WP      50
#define CCH     192              // Cin + Cout
#define NGATE   512              // 4*Cout
#define PIXHW   (HH*WW)          // 2304
#define NPIX    (BATCH*PIXHW)    // 18432
#define NIN     (NPIX*CIN)       // 1179648
#define NSTATE2 (NPIX*COUT)      // 2359296
#define SPITCH  104              // LDS row pitch (bf16 elems) to break bank conflicts

__device__ __forceinline__ float sigm(float v) { return 1.f / (1.f + expf(-v)); }

// Assemble a 16x32 bf16 WMMA A/B fragment from LDS (row-major [row][K]).
// Per ISA: lanes 0-15 hold K {0..7,16..23}, lanes 16-31 hold K {8..15,24..31};
// caller passes pointer already offset by klo = (lane>>4)*8.
__device__ __forceinline__ v16bf ld_frag(const __bf16* p) {
  union { v16bf f; uint4 u[2]; } r;
  r.u[0] = *(const uint4*)(p);        // K klo+0 .. klo+7
  r.u[1] = *(const uint4*)(p + 16);   // K klo+16 .. klo+23
  return r.f;
}

// ---------------- weight transform: OIHW fp32 -> [tap][N][K] bf16 ----------------
__global__ __launch_bounds__(256) void k_wt(const float* __restrict__ w, __bf16* __restrict__ wt) {
  int i = blockIdx.x * 256 + threadIdx.x;           // 9*512*192 = 884736
  if (i >= 9 * NGATE * CCH) return;
  int k   = i % CCH;
  int n   = (i / CCH) % NGATE;
  int tap = i / (CCH * NGATE);
  int ky = tap / 3, kx = tap % 3;
  wt[i] = (__bf16)w[((size_t)(n * CCH + k) * 3 + ky) * 3 + kx];
}

// ---------------- alpha neuron (elementwise) ----------------
__global__ __launch_bounds__(256) void k_alpha(const float* __restrict__ xt,
    float* __restrict__ syn_e, float* __restrict__ syn_i, float* __restrict__ spk1,
    __bf16* __restrict__ comb,
    const float* __restrict__ alpha_p, const float* __restrict__ beta_p,
    const float* __restrict__ th1_p) {
  int i = blockIdx.x * 256 + threadIdx.x;
  if (i >= NIN) return;
  float a = fminf(fmaxf(*alpha_p, 1e-6f), 1.f - 1e-6f);
  float b = fminf(fmaxf(*beta_p,  1e-6f), 1.f - 1e-6f);
  float la = logf(a), lb = logf(b);
  float tau = la / (lb - la);
  float th1 = *th1_p;
  int c = i & (CIN - 1);
  int p = i >> 6;                      // pixel index
  int bb  = p / PIXHW;
  int pix = p - bb * PIXHW;
  int y = pix / WW, x = pix - y * WW;
  float xv = xt[((size_t)(bb * CIN + c) * HH + y) * WW + x];
  float se = a * syn_e[i] + xv;
  float si = b * syn_i[i] - xv;
  float m1 = tau * (se + si) - spk1[i] * th1;
  float sp = (m1 - th1) > 0.f ? 1.f : 0.f;
  syn_e[i] = se; syn_i[i] = si; spk1[i] = sp;
  comb[((size_t)(bb * HP + y + 1) * WP + x + 1) * CCH + c] = (__bf16)sp;
}

// ---------------- implicit-GEMM conv via bf16 WMMA ----------------
// comb: [B][50][50][192] bf16 (zero halo). wt: [9][512][192] bf16. cc: [18432][512] f32.
__global__ __launch_bounds__(256) void k_gemm(const __bf16* __restrict__ comb,
                                              const __bf16* __restrict__ wt,
                                              float* __restrict__ cc) {
  __shared__ __bf16 sA[128][SPITCH];
  __shared__ __bf16 sB[128][SPITCH];
  const int tid  = threadIdx.x;
  const int lane = tid & 31;
  const int wave = tid >> 5;           // 0..7
  const int wm   = wave & 3;           // 4 row groups of 32
  const int wn   = wave >> 2;          // 2 col groups of 64
  const int mblk = blockIdx.x % 144;
  const int nblk = blockIdx.x / 144;
  const int m0 = mblk * 128;
  const int n0 = nblk * 128;

  v8f acc[2][4] = {};

  const int klo  = (lane >> 4) * 8;
  const int arow = wm * 32 + (lane & 15);
  const int brow = wn * 64 + (lane & 15);

  for (int tap = 0; tap < 9; ++tap) {
    const int ky = tap / 3, kx = tap % 3;
    for (int half = 0; half < 2; ++half) {
      __syncthreads();
      const int kofs = half * 96;
      // cooperative stage: 128 rows x 96 cols each for A and B (16B chunks)
      #pragma unroll
      for (int it = 0; it < 6; ++it) {
        int q  = tid + 256 * it;       // 0..1535
        int r  = q / 12;               // row 0..127
        int c8 = (q % 12) * 8;         // col 0..88 step 8
        int p   = m0 + r;
        int bb  = p / PIXHW;
        int pix = p - bb * PIXHW;
        int y = pix / WW, x = pix - y * WW;
        const __bf16* ga = comb + ((size_t)(bb * HP + (y + ky)) * WP + (x + kx)) * CCH + kofs + c8;
        *(uint4*)&sA[r][c8] = *(const uint4*)ga;
        const __bf16* gb = wt + ((size_t)tap * NGATE + n0 + r) * CCH + kofs + c8;
        *(uint4*)&sB[r][c8] = *(const uint4*)gb;
      }
      __syncthreads();
      #pragma unroll
      for (int ks = 0; ks < 3; ++ks) {
        const int kb = ks * 32 + klo;
        v16bf a0 = ld_frag(&sA[arow     ][kb]);
        v16bf a1 = ld_frag(&sA[arow + 16][kb]);
        v16bf b0 = ld_frag(&sB[brow     ][kb]);
        v16bf b1 = ld_frag(&sB[brow + 16][kb]);
        v16bf b2 = ld_frag(&sB[brow + 32][kb]);
        v16bf b3 = ld_frag(&sB[brow + 48][kb]);
        acc[0][0] = __builtin_amdgcn_wmma_f32_16x16x32_bf16(false, a0, false, b0, (short)0, acc[0][0], false, false);
        acc[0][1] = __builtin_amdgcn_wmma_f32_16x16x32_bf16(false, a0, false, b1, (short)0, acc[0][1], false, false);
        acc[0][2] = __builtin_amdgcn_wmma_f32_16x16x32_bf16(false, a0, false, b2, (short)0, acc[0][2], false, false);
        acc[0][3] = __builtin_amdgcn_wmma_f32_16x16x32_bf16(false, a0, false, b3, (short)0, acc[0][3], false, false);
        acc[1][0] = __builtin_amdgcn_wmma_f32_16x16x32_bf16(false, a1, false, b0, (short)0, acc[1][0], false, false);
        acc[1][1] = __builtin_amdgcn_wmma_f32_16x16x32_bf16(false, a1, false, b1, (short)0, acc[1][1], false, false);
        acc[1][2] = __builtin_amdgcn_wmma_f32_16x16x32_bf16(false, a1, false, b2, (short)0, acc[1][2], false, false);
        acc[1][3] = __builtin_amdgcn_wmma_f32_16x16x32_bf16(false, a1, false, b3, (short)0, acc[1][3], false, false);
      }
    }
  }
  // store: C/D layout -> lane L, vgpr g : row = g + 8*(L>>4), col = L&15
  const int rbase = m0 + wm * 32 + 8 * (lane >> 4);
  const int cbase = n0 + wn * 64 + (lane & 15);
  #pragma unroll
  for (int i = 0; i < 2; ++i)
    #pragma unroll
    for (int j = 0; j < 4; ++j)
      #pragma unroll
      for (int g = 0; g < 8; ++g)
        cc[(size_t)(rbase + i * 16 + g) * NGATE + cbase + j * 16] = acc[i][j][g];
}

// ---------------- LSTM gates + spike + per-channel spike count ----------------
__global__ __launch_bounds__(256) void k_gates(const float* __restrict__ cc,
    const float* __restrict__ b_conv, float* __restrict__ syn2, float* __restrict__ mem2,
    __bf16* __restrict__ comb, const float* __restrict__ th2_p, float* __restrict__ bn_sum) {
  const int tid = threadIdx.x;
  const int c   = tid & (COUT - 1);
  const float th2 = *th2_p;
  const float bi = b_conv[c], bf = b_conv[COUT + c], bg = b_conv[2 * COUT + c], bo = b_conv[3 * COUT + c];
  float cnt = 0.f;
  for (int p = blockIdx.x * 2 + (tid >> 7); p < NPIX; p += gridDim.x * 2) {
    const float* row = cc + (size_t)p * NGATE;
    float gi = sigm(row[c] + bi);
    float gf = sigm(row[COUT + c] + bf);
    float gg = tanhf(row[2 * COUT + c] + bg);
    float go = sigm(row[3 * COUT + c] + bo);
    size_t idx = (size_t)p * COUT + c;
    float s2 = gf * syn2[idx] + gi * gg;
    float m2 = go * tanhf(s2);
    syn2[idx] = s2; mem2[idx] = m2;
    cnt += (m2 - th2) > 0.f ? 1.f : 0.f;
    int bb = p / PIXHW, pix = p - bb * PIXHW;
    int y = pix / WW, x = pix - y * WW;
    comb[((size_t)(bb * HP + y + 1) * WP + x + 1) * CCH + CIN + c] = (__bf16)m2;
  }
  __shared__ float red[256];
  red[tid] = cnt;
  __syncthreads();
  if (tid < COUT) atomicAdd(&bn_sum[tid], red[tid] + red[tid + COUT]);
}

// ---------------- BN output (binary spikes: var = mu - mu^2) ----------------
__global__ __launch_bounds__(256) void k_bnout(const float* __restrict__ mem2,
    const float* __restrict__ bn_sum, const float* __restrict__ gamma,
    const float* __restrict__ bn_beta, const float* __restrict__ th2_p,
    float* __restrict__ out_t) {
  int i = blockIdx.x * 256 + threadIdx.x;          // NCHW over [B,128,48,48]
  if (i >= NSTATE2) return;
  int pix = i % PIXHW;
  int c   = (i / PIXHW) & (COUT - 1);
  int bb  = i / (PIXHW * COUT);
  float th2 = *th2_p;
  float mu  = bn_sum[c] * (1.f / (float)NPIX);
  float var = mu - mu * mu;
  float sp  = (mem2[(size_t)(bb * PIXHW + pix) * COUT + c] - th2) > 0.f ? 1.f : 0.f;
  out_t[i] = gamma[c] * (sp - mu) * rsqrtf(var + 1e-5f) + bn_beta[c];
}

extern "C" void kernel_launch(void* const* d_in, const int* in_sizes, int n_in,
                              void* d_out, int out_size, void* d_ws, size_t ws_size,
                              hipStream_t stream) {
  const float* x       = (const float*)d_in[0];
  const float* alpha   = (const float*)d_in[1];
  const float* beta    = (const float*)d_in[2];
  const float* th1     = (const float*)d_in[3];
  const float* th2     = (const float*)d_in[4];
  const float* w_conv  = (const float*)d_in[5];
  const float* b_conv  = (const float*)d_in[6];
  const float* gamma   = (const float*)d_in[7];
  const float* bn_beta = (const float*)d_in[8];
  float* out = (float*)d_out;

  char* ws = (char*)d_ws;
  __bf16* comb  = (__bf16*)ws;                                   // 8*50*50*192 bf16 = 7,680,000 B
  float* syn_e  = (float*)(ws + 7680000);                        // NIN f32
  float* syn_i  = syn_e + NIN;
  float* spk1   = syn_i + NIN;
  float* syn2   = spk1 + NIN;                                    // NSTATE2 f32
  float* mem2   = syn2 + NSTATE2;
  float* bn_sum = mem2 + NSTATE2;                                // T*128 f32
  __bf16* wt    = (__bf16*)(bn_sum + T_STEPS * COUT);            // 9*512*192 bf16
  float* cc     = (float*)((char*)wt + (size_t)9 * NGATE * CCH * 2);  // NPIX*512 f32

  // zero: comb halo + all recurrent state + bn_sum (contiguous prefix of ws)
  size_t zero_bytes = 7680000 + (size_t)(3 * NIN + 2 * NSTATE2 + T_STEPS * COUT) * 4;
  hipMemsetAsync(d_ws, 0, zero_bytes, stream);

  k_wt<<<(9 * NGATE * CCH + 255) / 256, 256, 0, stream>>>(w_conv, wt);

  for (int t = 0; t < T_STEPS; ++t) {
    k_alpha<<<NIN / 256, 256, 0, stream>>>(x + (size_t)t * NIN, syn_e, syn_i, spk1,
                                           comb, alpha, beta, th1);
    k_gemm<<<576, 256, 0, stream>>>(comb, wt, cc);
    k_gates<<<512, 256, 0, stream>>>(cc, b_conv, syn2, mem2, comb, th2,
                                     bn_sum + t * COUT);
    k_bnout<<<NSTATE2 / 256, 256, 0, stream>>>(mem2, bn_sum + t * COUT, gamma, bn_beta,
                                               th2, out + (size_t)t * NSTATE2);
  }
}
